// V1ExpectationModel_41609643164293
// MI455X (gfx1250) — compile-verified, matrix operand order checked by reading, not run on hardware
//
#include <hip/hip_runtime.h>

// ---------------------------------------------------------------------------
// V1 expectation scan for MI455X (gfx1250, wave32).
// Persistent recurrent state in LDS; bank mixing via v_wmma_f32_16x16x32_bf16;
// per-step logits double-buffered with global_load_async_to_lds_b128 (ASYNCcnt).
// ---------------------------------------------------------------------------

typedef __attribute__((ext_vector_type(16))) __bf16 v16bf;
typedef __attribute__((ext_vector_type(8)))  float  v8f;

#define N_ORI 180
#define NP    192              // padded: 6 K-tiles of 32 / 12 N-tiles of 16
#define BATCH 512
#define SEQ   256
#define BM    16               // batch rows per block == WMMA M
#define NWAVE 12               // one wave per 16-wide N tile
#define NTHREADS (NWAVE * 32)
#define BANK_ELEMS (6 * 12 * 32 * 16)   // 36864 bf16

// LDS partition (bytes)
#define OFF_BANK   0u
#define OFF_TMPLA  73728u                 // bank: 36864*2
#define OFF_STATE  79872u                 // tmplA: 3072*2
#define OFF_RED    141312u                // state: 5*16*192*4
#define OFF_STAGE  141568u                // red: 4*16*4
#define STAGE_BYTES 11520u                // 16*180*4 per buffer
#define SMEM_TOTAL (OFF_STAGE + 2u * STAGE_BYTES)   // 164608

__device__ __forceinline__ unsigned short f32_to_bf16_rne(float f) {
    union { float f; unsigned u; } x; x.f = f;
    unsigned r = x.u + 0x7FFFu + ((x.u >> 16) & 1u);
    return (unsigned short)(r >> 16);
}

// Async 16B copy global->LDS (GVS mode: saddr base + 32-bit vgpr offset).
__device__ __forceinline__ void async_load_b128(unsigned ldsAddr, unsigned gOff,
                                                const void* base) {
    asm volatile("global_load_async_to_lds_b128 %0, %1, %2"
                 :: "v"(ldsAddr), "v"(gOff), "s"(base)
                 : "memory");
}

// Stage one step's logits tile (16 rows x 180 f32 = 720 16B chunks) into LDS.
// Wave w covers chunks [w*60, w*60+60): 32 lanes then 28 lanes -> exactly two
// async instructions per wave per step (uniform ASYNCcnt bookkeeping).
__device__ __forceinline__ void issue_logits_stage(const float* logits, int s, int b0,
                                                   unsigned stageBase, int wave, int lane) {
    {
        int c = wave * 60 + lane;                      // all 32 lanes valid (60 > 32)
        int r = c / 45, q = c - r * 45;
        unsigned gOff = (unsigned)((b0 + r) * SEQ + s) * 720u + (unsigned)q * 16u;
        async_load_b128(stageBase + (unsigned)c * 16u, gOff, logits);
    }
    if (lane < 28) {                                   // partial EXEC, still 1 instr/wave
        int c = wave * 60 + 32 + lane;
        int r = c / 45, q = c - r * 45;
        unsigned gOff = (unsigned)((b0 + r) * SEQ + s) * 720u + (unsigned)q * 16u;
        async_load_b128(stageBase + (unsigned)c * 16u, gOff, logits);
    }
}

// ---------------------------------------------------------------------------
// Prep: fold banks (broad - 0.5*narrow) -> bf16 in WMMA B-fragment lane order
// [kt][nt][lane][elem]:  base = (i<8)?i:16+(i-8); k = kt*32 + base + (lane/16)*8
// ---------------------------------------------------------------------------
__global__ void prep_bank_kernel(const float* __restrict__ narrow,
                                 const float* __restrict__ broad,
                                 unsigned short* __restrict__ bankB_g)
{
    int idx = blockIdx.x * blockDim.x + threadIdx.x;
    if (idx >= BANK_ELEMS) return;
    int i    = idx & 15;
    int lane = (idx >> 4) & 31;
    int nt   = (idx >> 9) % 12;
    int kt   = idx / (16 * 32 * 12);
    int g    = lane >> 4;
    int ncol = nt * 16 + (lane & 15);
    int base = (i < 8) ? i : (16 + (i - 8));
    int k    = kt * 32 + base + g * 8;
    float val = 0.0f;
    if (k < N_ORI && ncol < N_ORI)
        val = broad[k * N_ORI + ncol] - 0.5f * narrow[k * N_ORI + ncol];
    bankB_g[idx] = f32_to_bf16_rne(val);
}

// ---------------------------------------------------------------------------
// Main: one block = 16 batch rows, all 256 steps. State lives in LDS.
// ---------------------------------------------------------------------------
__global__ void __launch_bounds__(NTHREADS)
v1_scan_kernel(const int*   __restrict__ ori,
               const float* __restrict__ logits,
               const float* __restrict__ prec_logit,
               const unsigned short* __restrict__ bankB_g,
               float* __restrict__ out)
{
    extern __shared__ char smem_raw[];
    unsigned short* bankB = (unsigned short*)(smem_raw + OFF_BANK);
    unsigned short* tmplA = (unsigned short*)(smem_raw + OFF_TMPLA);
    float* l4F   = (float*)(smem_raw + OFF_STATE);
    float* l23F  = l4F   + BM * NP;
    float* tmplF = l23F  + BM * NP;
    float* adF   = tmplF + BM * NP;
    float* compF = adF   + BM * NP;
    float* driveSum = (float*)(smem_raw + OFF_RED);   // [16]
    float* expSum   = driveSum + BM;                  // [16]
    float* precS    = expSum   + BM;                  // [16]
    int*   oriRow   = (int*)(precS + BM);             // [16]
    float* stage0   = (float*)(smem_raw + OFF_STAGE); // [16*180]
    float* stage1   = stage0 + BM * N_ORI;

    const int tid  = threadIdx.x;
    const int b0   = blockIdx.x * BM;
    const int wave = tid >> 5;
    const int lane = tid & 31;
    const int colT = tid % 192;      // fixed column (once; no per-iter div)
    const int rhT  = tid / 192;      // row-pair half (0/1)

    const unsigned stageAddr[2] = { (unsigned)(size_t)stage0, (unsigned)(size_t)stage1 };
    float* const stagePtr[2] = { stage0, stage1 };

    // Kick off async stage of step 0's logits immediately (overlaps init).
    issue_logits_stage(logits, 0, b0, stageAddr[0], wave, lane);

    // Stage constant bank into LDS (73728/16 = 4608 uint4 chunks).
    {
        const uint4* src = (const uint4*)bankB_g;
        uint4*       dst = (uint4*)bankB;
        for (int t = tid; t < 4608; t += NTHREADS) dst[t] = src[t];
    }
    // Zero recurrent state (5 contiguous planes) and padded A fragments.
    for (int t = tid; t < 5 * BM * NP; t += NTHREADS) l4F[t] = 0.0f;
    for (int t = tid; t < BM * NP / 2; t += NTHREADS) tmplA[t] = 0;
    __syncthreads();

    const size_t plane = (size_t)BATCH * SEQ * N_ORI;

    for (int s = 0; s < SEQ; ++s) {
        const float* stageCur = stagePtr[s & 1];

        // ---- phase 0: per-row scalars; issue next stage; drain current ----
        if (tid < BM) {
            int r = tid, b = b0 + r;
            driveSum[r] = 0.0f;
            expSum[r]   = 0.0f;
            oriRow[r]   = ori[b * SEQ + s];
            float pl    = prec_logit[b * SEQ + s];
            precS[r]    = 1.0f / (1.0f + __expf(-pl));
        }
        if (s + 1 < SEQ) {
            issue_logits_stage(logits, s + 1, b0, stageAddr[(s + 1) & 1], wave, lane);
            asm volatile("s_wait_asynccnt 0x2" ::: "memory");  // current 2 done, next 2 in flight
        } else {
            asm volatile("s_wait_asynccnt 0x0" ::: "memory");
        }
        __syncthreads();

        // ---- phase 1: adaptation EMA, drive, wave-reduced row sums ----
        #pragma unroll
        for (int j = 0; j < 8; ++j) {
            int r = (j << 1) | rhT;
            int o = r * NP + colT;
            float drive = 0.0f, e = 0.0f;
            if (colT < N_ORI) {
                float sens  = (colT == oriRow[r]) ? 1.0f : 0.0f;
                float ad    = 0.9f * adF[o] + 0.2f * l4F[o];
                adF[o] = ad;
                drive = fmaxf(sens - ad, 0.0f);
                l4F[o] = drive;                          // stash drive (old l4 consumed)
                e = __expf(stageCur[r * N_ORI + colT]);
                compF[o] = e;                            // stash exp (comp free until ph3)
            }
            float dsum = drive, esum = e;                // butterfly over wave (all same r)
            #pragma unroll
            for (int m = 16; m >= 1; m >>= 1) {
                dsum += __shfl_xor(dsum, m);
                esum += __shfl_xor(esum, m);
            }
            if (lane == 0) {
                atomicAdd(&driveSum[r], dsum);
                atomicAdd(&expSum[r], esum);
            }
        }
        __syncthreads();

        // ---- phase 2: divisive norm, template EMA, emit bf16 A fragments ----
        #pragma unroll
        for (int j = 0; j < 8; ++j) {
            int r = (j << 1) | rhT;
            if (colT < N_ORI) {
                int o = r * NP + colT;
                float l4 = l4F[o] / (1.0f + driveSum[r] * (1.0f / N_ORI));
                l4F[o] = l4;
                float tm = 0.8f * tmplF[o] + 0.2f * precS[r] * (compF[o] / expSum[r]);
                tmplF[o] = tm;
                // scatter into A-fragment layout: lane = g*16 + r, elem i
                int kt = colT >> 5, kin = colT & 31;
                int g  = (kin >> 3) & 1;
                int base = kin - 8 * g;                  // {0..7} or {16..23}
                int i  = (base & 7) + ((base >> 4) << 3);
                tmplA[((kt * 32) + g * 16 + r) * 16 + i] = f32_to_bf16_rne(tm);
            }
        }
        __syncthreads();

        // ---- phase 3: comparator_raw = template @ (broad - 0.5*narrow) ----
        {
            const int nt = wave;                         // 12 waves = 12 N tiles
            v8f c = {0.f, 0.f, 0.f, 0.f, 0.f, 0.f, 0.f, 0.f};
            #pragma unroll
            for (int kt = 0; kt < 6; ++kt) {
                const v16bf a = *(const v16bf*)&tmplA[(kt * 32 + lane) * 16];
                const v16bf b = *(const v16bf*)&bankB[((kt * 12 + nt) * 32 + lane) * 16];
                c = __builtin_amdgcn_wmma_f32_16x16x32_bf16(
                        /*neg_a=*/false, a, /*neg_b=*/false, b,
                        /*c_mod=*/(short)0, c, /*reuse_a=*/false, /*reuse_b=*/false);
            }
            // D layout: lanes 0-15 -> M=v, lanes 16-31 -> M=v+8; N = nt*16 + (lane&15)
            int m0 = (lane >> 4) * 8;
            int nc = nt * 16 + (lane & 15);
            #pragma unroll
            for (int v = 0; v < 8; ++v)
                compF[(m0 + v) * NP + nc] = c[v];
        }
        __syncthreads();

        // ---- phase 4: l23 update + stream 5 state planes to global ----
        #pragma unroll
        for (int j = 0; j < 8; ++j) {
            int r = (j << 1) | rhT;
            if (colT < N_ORI) {
                int o = r * NP + colT;
                float comparator = precS[r] * compF[o];
                float l23 = fmaxf(l4F[o] + 0.5f * l23F[o] + tmplF[o] - comparator, 0.0f);
                l23F[o] = l23;
                size_t ob = ((size_t)(b0 + r) * SEQ + s) * N_ORI + colT;
                out[ob]             = l4F[o];
                out[ob + plane]     = l23;
                out[ob + 2 * plane] = tmplF[o];
                out[ob + 3 * plane] = adF[o];
                out[ob + 4 * plane] = comparator;
            }
        }
        __syncthreads();
    }
}

extern "C" void kernel_launch(void* const* d_in, const int* in_sizes, int n_in,
                              void* d_out, int out_size, void* d_ws, size_t ws_size,
                              hipStream_t stream)
{
    const int*   orientations = (const int*)d_in[0];    // (B,S) int32
    const float* logits       = (const float*)d_in[1];  // (B,S,N) f32
    const float* prec_lg      = (const float*)d_in[2];  // (B,S,1) f32
    const float* narrow       = (const float*)d_in[3];  // (N,N) f32
    const float* broad        = (const float*)d_in[4];  // (N,N) f32
    float*       out          = (float*)d_out;          // (5,B,S,N) f32
    unsigned short* bankB     = (unsigned short*)d_ws;  // 73728 B scratch

    prep_bank_kernel<<<(BANK_ELEMS + 255) / 256, 256, 0, stream>>>(narrow, broad, bankB);

    v1_scan_kernel<<<BATCH / BM, NTHREADS, SMEM_TOTAL, stream>>>(
        orientations, logits, prec_lg, bankB, out);
}
